// NetworkModel_27754078666826
// MI455X (gfx1250) — matrix operand order repaired and checked by reading, not compiled, not run on hardware
//
#include <hip/hip_runtime.h>
#include <hip/hip_bf16.h>

typedef __attribute__((ext_vector_type(16))) __bf16 v16bf;
typedef __attribute__((ext_vector_type(8)))  float  v8f;

#define NH 3
#define BSZ 2048
#define NROWS (NH * BSZ)     // 6144
#define DIN 4586
#define HID 256

// ---------------------------------------------------------------------------
// Weight repack: f32 [K,N] row-major -> bf16 fragment-ordered, zero padded.
// Layout: o = (((kt*NG + ng)*4 + j)*32 + lane)*16 + e
//   k = kt*32 + (e>>3)*16 + (lane>>4)*8 + (e&7) ; n = ng*64 + j*16 + (lane&15)
// ---------------------------------------------------------------------------
__global__ void __launch_bounds__(256) repack_w(
    const float* __restrict__ W, __bf16* __restrict__ Wpk,
    int K, int N, int NG, long long total)
{
    long long o = (long long)blockIdx.x * 256 + threadIdx.x;
    if (o >= total) return;
    int e    = (int)(o & 15);
    int lane = (int)((o >> 4) & 31);
    int j    = (int)((o >> 9) & 3);
    long long rest = o >> 11;
    int ng = (int)(rest % NG);
    int kt = (int)(rest / NG);
    int k = kt * 32 + (e >> 3) * 16 + (lane >> 4) * 8 + (e & 7);
    int n = ng * 64 + j * 16 + (lane & 15);
    float v = (k < K && n < N) ? W[(long long)k * N + n] : 0.0f;
    Wpk[o] = (__bf16)v;
}

// ---------------------------------------------------------------------------
// WMMA GEMM:  D[M,N] = act( A[M,K] @ W[K,N] + bias ), W pre-packed bf16.
// One wave32 computes a 16x64 output strip (4 accumulators, NT=4 blocking).
// A row addressing: row r -> A + (r/rowsPerGroup)*outer + (r%rowsPerGroup)*inner
// ---------------------------------------------------------------------------
__global__ void __launch_bounds__(256) gemm_bias_act(
    const float* __restrict__ A, long long outerStride, int rowsPerGroup, int innerStride,
    const __bf16* __restrict__ Wpk, const float* __restrict__ bias,
    float* __restrict__ D, int ldd,
    int M, int N, int K, int doRelu, int totalTiles, int NG, int alignedA)
{
    int wave = threadIdx.x >> 5;
    int lane = threadIdx.x & 31;
    int tile = blockIdx.x * 8 + wave;
    if (tile >= totalTiles) return;          // wave-uniform
    int tm = tile / NG;
    int ng = tile - tm * NG;

    int half = lane >> 4;
    int l16  = lane & 15;

    // clamped A row: out-of-range rows compute garbage that is never stored
    int mRow = tm * 16 + l16;
    int mC   = mRow < M ? mRow : (M - 1);
    int grp  = mC / rowsPerGroup;
    int ent  = mC - grp * rowsPerGroup;
    const float* Arow = A + (long long)grp * outerStride + (long long)ent * innerStride;

    const v16bf* Bp = (const v16bf*)Wpk;
    v8f acc[4] = {};

    int kFull = K & ~31;
    int kt = 0;

    if (alignedA) {
        for (int k0 = 0; k0 < kFull; k0 += 32, ++kt) {
            const float* p = Arow + k0 + half * 8;
            float4 f0 = *(const float4*)(p);
            float4 f1 = *(const float4*)(p + 4);
            float4 f2 = *(const float4*)(p + 16);
            float4 f3 = *(const float4*)(p + 20);
            v16bf a;
            a[0]=(__bf16)f0.x;  a[1]=(__bf16)f0.y;  a[2]=(__bf16)f0.z;  a[3]=(__bf16)f0.w;
            a[4]=(__bf16)f1.x;  a[5]=(__bf16)f1.y;  a[6]=(__bf16)f1.z;  a[7]=(__bf16)f1.w;
            a[8]=(__bf16)f2.x;  a[9]=(__bf16)f2.y;  a[10]=(__bf16)f2.z; a[11]=(__bf16)f2.w;
            a[12]=(__bf16)f3.x; a[13]=(__bf16)f3.y; a[14]=(__bf16)f3.z; a[15]=(__bf16)f3.w;
            long long bb = (((long long)kt * NG + ng) * 4) * 32 + lane;
            #pragma unroll
            for (int j = 0; j < 4; ++j) {
                v16bf b = Bp[bb + (long long)j * 32];
                acc[j] = __builtin_amdgcn_wmma_f32_16x16x32_bf16(
                             false, a, false, b, (short)0, acc[j], false, false);
            }
        }
    } else {
        for (int k0 = 0; k0 < kFull; k0 += 32, ++kt) {
            v16bf a;
            #pragma unroll
            for (int e = 0; e < 16; ++e) {
                int kk = k0 + ((e < 8) ? (half * 8 + e) : (16 + half * 8 + (e - 8)));
                a[e] = (__bf16)Arow[kk];         // unconditional, in-bounds (kk < kFull <= K)
            }
            long long bb = (((long long)kt * NG + ng) * 4) * 32 + lane;
            #pragma unroll
            for (int j = 0; j < 4; ++j) {
                v16bf b = Bp[bb + (long long)j * 32];
                acc[j] = __builtin_amdgcn_wmma_f32_16x16x32_bf16(
                             false, a, false, b, (short)0, acc[j], false, false);
            }
        }
    }

    if (kFull < K) {                          // K tail, once: clamped addr + select
        v16bf a;
        #pragma unroll
        for (int e = 0; e < 16; ++e) {
            int kk = kFull + ((e < 8) ? (half * 8 + e) : (16 + half * 8 + (e - 8)));
            int kc = kk < K ? kk : (K - 1);
            float v = Arow[kc];
            a[e] = (__bf16)(kk < K ? v : 0.0f);
        }
        long long bb = (((long long)kt * NG + ng) * 4) * 32 + lane;  // packed is zero-padded
        #pragma unroll
        for (int j = 0; j < 4; ++j) {
            v16bf b = Bp[bb + (long long)j * 32];
            acc[j] = __builtin_amdgcn_wmma_f32_16x16x32_bf16(
                         false, a, false, b, (short)0, acc[j], false, false);
        }
    }

    // C/D layout: lane -> col = l16 (of sub-tile j), VGPR r -> row = tm*16 + half*8 + r
    #pragma unroll
    for (int j = 0; j < 4; ++j) {
        int nCol = ng * 64 + j * 16 + l16;
        if (nCol < N) {
            float bv = bias[nCol];
            #pragma unroll
            for (int r = 0; r < 8; ++r) {
                int mo = tm * 16 + half * 8 + r;
                if (mo < M) {
                    float v = acc[j][r] + bv;
                    if (doRelu) v = fmaxf(v, 0.0f);
                    D[(long long)mo * ldd + nCol] = v;
                }
            }
        }
    }
}

// ---------------------------------------------------------------------------
// Conv stem: per-sample in LDS. 6x17x17 -> conv1(18,3x3,SAME)+ReLU
//            -> conv2(12,3x3,SAME)+ReLU -> [12*17*17] flattened out.
// ---------------------------------------------------------------------------
__global__ void __launch_bounds__(256) conv_stem(
    const float* __restrict__ heroes,
    const float* __restrict__ c1w, const float* __restrict__ c1b,
    const float* __restrict__ c2w, const float* __restrict__ c2b,
    float* __restrict__ conv_out)
{
    __shared__ float in_s[6 * 289];
    __shared__ float h1_s[18 * 289];
    int i = blockIdx.x;
    const float* src = heroes + (long long)i * DIN;
    for (int t = threadIdx.x; t < 6 * 289; t += 256) in_s[t] = src[t];
    __syncthreads();
    for (int t = threadIdx.x; t < 18 * 289; t += 256) {
        int oc = t / 289, p = t - oc * 289, y = p / 17, x = p - y * 17;
        float s = c1b[oc];
        for (int ic = 0; ic < 6; ++ic)
            for (int ky = 0; ky < 3; ++ky) {
                int yy = y + ky - 1; if (yy < 0 || yy > 16) continue;
                for (int kx = 0; kx < 3; ++kx) {
                    int xx = x + kx - 1; if (xx < 0 || xx > 16) continue;
                    s += in_s[ic * 289 + yy * 17 + xx] * c1w[((oc * 6 + ic) * 3 + ky) * 3 + kx];
                }
            }
        h1_s[t] = fmaxf(s, 0.0f);
    }
    __syncthreads();
    float* dst = conv_out + (long long)i * 3468;
    for (int t = threadIdx.x; t < 12 * 289; t += 256) {
        int oc = t / 289, p = t - oc * 289, y = p / 17, x = p - y * 17;
        float s = c2b[oc];
        for (int ic = 0; ic < 18; ++ic)
            for (int ky = 0; ky < 3; ++ky) {
                int yy = y + ky - 1; if (yy < 0 || yy > 16) continue;
                for (int kx = 0; kx < 3; ++kx) {
                    int xx = x + kx - 1; if (xx < 0 || xx > 16) continue;
                    s += h1_s[ic * 289 + yy * 17 + xx] * c2w[((oc * 18 + ic) * 3 + ky) * 3 + kx];
                }
            }
        dst[t] = fmaxf(s, 0.0f);
    }
}

// ---------------------------------------------------------------------------
// Max-pool entity tokens + assemble cat row [512 img | 9 x 64 pooled]
// ---------------------------------------------------------------------------
__global__ void __launch_bounds__(256) pool_cat(
    const float* __restrict__ imgh,
    const float* __restrict__ hf, const float* __restrict__ he,
    const float* __restrict__ pu,
    const float* __restrict__ sf, const float* __restrict__ se,
    const float* __restrict__ of_, const float* __restrict__ oe,
    const float* __restrict__ mo, const float* __restrict__ gl,
    float* __restrict__ cat)
{
    int i = blockIdx.x;
    float* out = cat + (long long)i * 1088;
    for (int c = threadIdx.x; c < 512; c += 256)
        out[c] = imgh[(long long)i * 512 + c];
    const float* bufs[9] = { hf, he, pu, sf, se, of_, oe, mo, gl };
    const int    cnt[9]  = { 3,  3,  1,  10, 10, 3,   3,  20, 1 };
    for (int t = threadIdx.x; t < 9 * 64; t += 256) {
        int seg = t >> 6, c = t & 63;
        const float* p = bufs[seg] + (long long)i * cnt[seg] * 64 + c;
        float v = p[0];
        for (int e = 1; e < cnt[seg]; ++e) v = fmaxf(v, p[(long long)e * 64]);
        out[512 + seg * 64 + c] = v;
    }
}

// ---------------------------------------------------------------------------
// Cross-hero mix: fc[h,b,:128] = max_h' ch[h',b,:128]; fc[...,128:] = ch
// ---------------------------------------------------------------------------
__global__ void __launch_bounds__(256) fc_mix(
    const float* __restrict__ ch, float* __restrict__ fc)
{
    long long idx = (long long)blockIdx.x * 256 + threadIdx.x;
    const long long total = (long long)NROWS * 512;
    if (idx >= total) return;
    int col = (int)(idx & 511);
    long long r = idx >> 9;               // r = h*B + b
    long long b = r % BSZ;
    float v;
    if (col < 128) {
        v =          ch[(b            ) * 512 + col];
        v = fmaxf(v, ch[(b + 1LL*BSZ) * 512 + col]);
        v = fmaxf(v, ch[(b + 2LL*BSZ) * 512 + col]);
    } else {
        v = ch[r * 512 + col];
    }
    fc[idx] = v;
}

// ---------------------------------------------------------------------------
// Final logits. Algebra: (t @ Wt + bt) . k == t . (Wt k) + bt . k
// Per sample: u = temb_w @ tkey (64x64 matvec), then 39 dot-64 products.
// ---------------------------------------------------------------------------
__global__ void __launch_bounds__(64) final_logits(
    const float* __restrict__ tkey,
    const float* __restrict__ temb_w, const float* __restrict__ temb_b,
    const float* __restrict__ hf, const float* __restrict__ he,
    const float* __restrict__ pu, const float* __restrict__ mo,
    const float* __restrict__ se, const float* __restrict__ oe,
    float* __restrict__ out)
{
    __shared__ float tk[64], u[64];
    int i = blockIdx.x;
    int tid = threadIdx.x;
    tk[tid] = tkey[(long long)i * 64 + tid];
    __syncthreads();
    float s = 0.0f;
    for (int j = 0; j < 64; ++j) s += temb_w[tid * 64 + j] * tk[j];
    u[tid] = s;
    __syncthreads();
    float cb = 0.0f, usum = 0.0f;
    for (int j = 0; j < 64; ++j) { cb += temb_b[j] * tk[j]; usum += u[j]; }
    if (tid < 39) {
        float lg = cb;
        if (tid == 0) {
            lg = 0.1f * usum + cb;
        } else if (tid <= 3) {
            const float* r = hf + ((long long)i * 3 + (tid - 1)) * 64;
            for (int j = 0; j < 64; ++j) lg += r[j] * u[j];
        } else if (tid <= 6) {
            const float* r = he + ((long long)i * 3 + (tid - 4)) * 64;
            for (int j = 0; j < 64; ++j) lg += r[j] * u[j];
        } else if (tid == 7) {
            const float* r = pu + (long long)i * 64;
            for (int j = 0; j < 64; ++j) lg += r[j] * u[j];
        } else if (tid <= 27) {
            const float* r = mo + ((long long)i * 20 + (tid - 8)) * 64;
            for (int j = 0; j < 64; ++j) lg += r[j] * u[j];
        } else if (tid <= 37) {
            const float* r = se + ((long long)i * 10 + (tid - 28)) * 64;
            for (int j = 0; j < 64; ++j) lg += r[j] * u[j];
        } else {
            const float* r = oe + (long long)i * 3 * 64;
            for (int j = 0; j < 64; ++j) {
                float m = fmaxf(fmaxf(r[j], r[64 + j]), r[128 + j]);
                lg += m * u[j];
            }
        }
        out[(long long)i * 119 + 80 + tid] = lg;
    }
}

// ---------------------------------------------------------------------------
static __bf16* pack_weight(hipStream_t s, const float* W, int K, int N, __bf16** pkCur)
{
    int kSteps = (K + 31) / 32, NG = (N + 63) / 64;
    long long total = (long long)kSteps * NG * 2048;   // *4*32*16
    __bf16* dst = *pkCur;
    *pkCur += total;
    repack_w<<<(int)((total + 255) / 256), 256, 0, s>>>(W, dst, K, N, NG, total);
    return dst;
}

static void launch_gemm(hipStream_t s,
                        const float* A, long long outer, int rpg, int inner,
                        const __bf16* Wpk, const float* bias, float* D, int ldd,
                        int M, int N, int K, int relu, int alignedA)
{
    int tilesM = (M + 15) / 16, NG = (N + 63) / 64;
    int total = tilesM * NG;
    int blocks = (total + 7) / 8;
    gemm_bias_act<<<blocks, 256, 0, s>>>(A, outer, rpg, inner, Wpk, bias, D, ldd,
                                         M, N, K, relu, total, NG, alignedA);
}

extern "C" void kernel_launch(void* const* d_in, const int* in_sizes, int n_in,
                              void* d_out, int out_size, void* d_ws, size_t ws_size,
                              hipStream_t stream)
{
    // ---- input ordering: dict-insertion vs JAX pytree (sorted keys) --------
    bool insertion = (n_in > 1 && in_sizes[1] == 972);
    static const int sortedMap[52] = {
        8, 7, 10, 9, 24, 23,
        19, 20, 21, 22,   15, 16, 17, 18,
        45, 46, 47, 48,   41, 42, 43, 44,
        33, 34, 35, 36,   29, 30, 31, 32,
        25, 26, 27, 28,   37, 38, 39, 40,
        11, 12, 13, 14,
        6, 5,   4, 3,   2, 1,   52, 51,   50, 49
    };
    auto PRM = [&](int k) -> const float* {
        return (const float*)d_in[insertion ? (1 + k) : sortedMap[k]];
    };
    const float* heroes = (const float*)d_in[0];
    float* out = (float*)d_out;

    const float* conv1_w = PRM(0);  const float* conv1_b = PRM(1);
    const float* conv2_w = PRM(2);  const float* conv2_b = PRM(3);
    const float* img_w   = PRM(4);  const float* img_b   = PRM(5);
    struct MLP { const float *w1, *b1, *w2, *b2; };
    MLP mlp[9];
    for (int m = 0; m < 9; ++m) {
        mlp[m].w1 = PRM(6 + m * 4 + 0); mlp[m].b1 = PRM(6 + m * 4 + 1);
        mlp[m].w2 = PRM(6 + m * 4 + 2); mlp[m].b2 = PRM(6 + m * 4 + 3);
    }
    const float* concat_w = PRM(42); const float* concat_b = PRM(43);
    const float* comm_w   = PRM(44); const float* comm_b   = PRM(45);
    const float* action_w = PRM(46); const float* action_b = PRM(47);
    const float* tkey_w   = PRM(48); const float* tkey_b   = PRM(49);
    const float* temb_w   = PRM(50); const float* temb_b   = PRM(51);

    // ---- workspace carve-up ------------------------------------------------
    float* w = (float*)d_ws;
    long long off = 0;
    auto alloc = [&](long long nfl) { float* p = w + off; off += nfl; return p; };
    const long long n = NROWS;
    float* conv_out = alloc(n * 3468);
    float* imgh     = alloc(n * 512);
    float* hidden   = alloc(n * 20 * HID);   // max entity rows = n*20
    float* hf  = alloc(n * 3 * 64);
    float* he  = alloc(n * 3 * 64);
    float* sf  = alloc(n * 10 * 64);
    float* se  = alloc(n * 10 * 64);
    float* of_ = alloc(n * 3 * 64);
    float* oe  = alloc(n * 3 * 64);
    float* mo  = alloc(n * 20 * 64);
    float* pu  = alloc(n * 64);
    float* gl  = alloc(n * 64);
    float* cat = alloc(n * 1088);
    float* ch  = alloc(n * 512);
    float* fc  = alloc(n * 512);
    float* comm = alloc(n * 512);
    float* tkey = alloc(n * 64);
    __bf16* pkCur = (__bf16*)(w + off);      // bf16 packed-weight arena
    (void)ws_size; (void)out_size;

    // ---- 0) pack all weights to fragment-ordered bf16 ---------------------
    __bf16* img_pk    = pack_weight(stream, img_w,    3468, 512, &pkCur);
    __bf16* w1_pk[9]; __bf16* w2_pk[9];
    const int feats[9] = { 251, 251, 25, 25, 29, 29, 28, 44, 68 };
    for (int m = 0; m < 9; ++m) {
        w1_pk[m] = pack_weight(stream, mlp[m].w1, feats[m], HID, &pkCur);
        w2_pk[m] = pack_weight(stream, mlp[m].w2, HID, 64, &pkCur);
    }
    __bf16* concat_pk = pack_weight(stream, concat_w, 1088, 512, &pkCur);
    __bf16* comm_pk   = pack_weight(stream, comm_w,   512,  512, &pkCur);
    __bf16* action_pk = pack_weight(stream, action_w, 512,  80,  &pkCur);
    __bf16* tkey_pk   = pack_weight(stream, tkey_w,   512,  64,  &pkCur);

    // ---- 1) conv stem + image FC ------------------------------------------
    conv_stem<<<(int)n, 256, 0, stream>>>(heroes, conv1_w, conv1_b, conv2_w, conv2_b, conv_out);
    launch_gemm(stream, conv_out, 3468, 1, 0, img_pk, img_b, imgh, 512,
                (int)n, 512, 3468, 1, 1);

    // ---- 2) entity MLPs (grouped-row GEMMs straight out of heroes) --------
    struct Ent { int offset, E, F; float* dst; };
    const Ent ents[9] = {
        { 1734, 3, 251, hf }, { 2487, 3, 251, he },
        { 3284, 10, 25, sf }, { 3534, 10, 25, se },
        { 3784, 3, 29, of_ }, { 3871, 3, 29, oe },
        { 3958, 20, 28, mo }, { 3240, 1, 44, pu }, { 4518, 1, 68, gl }
    };
    for (int m = 0; m < 9; ++m) {
        int M = (int)n * ents[m].E;
        launch_gemm(stream, heroes + ents[m].offset, DIN, ents[m].E, ents[m].F,
                    w1_pk[m], mlp[m].b1, hidden, HID, M, HID, ents[m].F, 1, /*aligned*/0);
        launch_gemm(stream, hidden, HID, 1, 0,
                    w2_pk[m], mlp[m].b2, ents[m].dst, 64, M, 64, HID, 1, 1);
    }

    // ---- 3) pool + concat -> concat FC -> cross-hero mix -> comm FC -------
    pool_cat<<<(int)n, 256, 0, stream>>>(imgh, hf, he, pu, sf, se, of_, oe, mo, gl, cat);
    launch_gemm(stream, cat, 1088, 1, 0, concat_pk, concat_b, ch, 512,
                (int)n, 512, 1088, 1, 1);
    {
        long long tot = n * 512;
        fc_mix<<<(int)((tot + 255) / 256), 256, 0, stream>>>(ch, fc);
    }
    launch_gemm(stream, fc, 512, 1, 0, comm_pk, comm_b, comm, 512,
                (int)n, 512, 512, 1, 1);

    // ---- 4) heads ----------------------------------------------------------
    launch_gemm(stream, comm, 512, 1, 0, action_pk, action_b, out, 119,
                (int)n, 80, 512, 0, 1);
    launch_gemm(stream, comm, 512, 1, 0, tkey_pk, tkey_b, tkey, 64,
                (int)n, 64, 512, 0, 1);
    final_logits<<<(int)n, 64, 0, stream>>>(tkey, temb_w, temb_b,
                                            hf, he, pu, mo, se, oe, out);
}
